// DeformableAttention_24163486007415
// MI455X (gfx1250) — matrix operand order repaired and checked by reading, not compiled, not run on hardware
//
#include <hip/hip_runtime.h>

#define DM   256
#define NHEADS 8
#define NPTS   4
#define HH   100
#define WW   100
#define BB     4
#define QQ   (HH * WW)
#define SS   (HH * WW)
#define HDIM  32                 // DM / NHEADS
#define MROWS (BB * QQ)          // 40000
#define NOA   96                 // 64 offset cols + 32 attn cols

typedef unsigned short u16;
typedef __attribute__((ext_vector_type(16))) __bf16 v16bf;
typedef __attribute__((ext_vector_type(8)))  __bf16 v8bf;
typedef __attribute__((ext_vector_type(8)))  float  v8f;

union FragA { v16bf v; v8bf h[2]; };

__device__ __forceinline__ u16 f2bf(float f) {
  unsigned u = __float_as_uint(f);
  u += 0x7FFFu + ((u >> 16) & 1u);           // round-to-nearest-even
  return (u16)(u >> 16);
}
__device__ __forceinline__ float bf2f(u16 h) {
  return __uint_as_float(((unsigned)h) << 16);
}

// ---------------------------------------------------------------- converts
__global__ __launch_bounds__(256) void convert_f32_bf16(
    const float* __restrict__ in, u16* __restrict__ out, int n) {
  int i = blockIdx.x * blockDim.x + threadIdx.x;
  int stride = gridDim.x * blockDim.x;
  for (; i < n; i += stride) out[i] = f2bf(in[i]);
}

__global__ __launch_bounds__(128) void pack_bias_96(
    const float* __restrict__ b_off, const float* __restrict__ b_attn,
    float* __restrict__ out) {
  int i = threadIdx.x;
  if (i < 64) out[i] = b_off[i];
  else if (i < 96) out[i] = b_attn[i - 64];
}

// ---------------------------------------------------------------- WMMA GEMM
// C[m,n] = sum_k A[m,k] * Wt[n,k] + bias[n]
// One wave per block; each wave owns a 16 x (16*TN) output strip.
// Register double-buffering: fragments for K-chunk k+1 are issued before the
// WMMAs of chunk k so global loads overlap the matrix pipe within the wave.
// MODE 0: f32 row-major [M,N] output.  MODE 1: bf16 output scattered into the
// value-gather layout [B, NH, S, 32].
template <int TN, int MODE>
__global__ __launch_bounds__(32) void gemm_bf16_wmma(
    const u16* __restrict__ A,      // [M, 256] bf16
    const u16* __restrict__ Wt,     // [N, 256] bf16
    const float* __restrict__ bias, // [N]
    float* __restrict__ Cf,         // MODE 0
    u16* __restrict__ Cb,           // MODE 1
    int N) {
  const int lane = threadIdx.x & 31;
  const int mt   = blockIdx.x;
  const int ntB  = blockIdx.y * TN;

  const int rowA = mt * 16 + (lane & 15);
  const int kgA  = (lane >> 4) << 4;   // A-frag: contiguous K {kg..kg+15}
  const int kgB  = (lane >> 4) << 4;   // B-frag: contiguous K {kb..kb+15}

  const u16* aRow = A + (size_t)rowA * DM + kgA;
  const u16* bRow[TN];
#pragma unroll
  for (int j = 0; j < TN; ++j)
    bRow[j] = Wt + (size_t)((ntB + j) * 16 + (lane & 15)) * DM + kgB;

  v8f acc[TN];
  const v8f zero = {0.f, 0.f, 0.f, 0.f, 0.f, 0.f, 0.f, 0.f};
#pragma unroll
  for (int j = 0; j < TN; ++j) acc[j] = zero;

  FragA a0, a1;
  v16bf b0[TN], b1[TN];

  // preload K-chunk 0
  a0.h[0] = *reinterpret_cast<const v8bf*>(aRow);
  a0.h[1] = *reinterpret_cast<const v8bf*>(aRow + 8);
#pragma unroll
  for (int j = 0; j < TN; ++j)
    b0[j] = *reinterpret_cast<const v16bf*>(bRow[j]);

#pragma unroll
  for (int k0 = 0; k0 < DM; k0 += 64) {
    // prefetch chunk k0+32 while chunk k0 computes
    a1.h[0] = *reinterpret_cast<const v8bf*>(aRow + k0 + 32);
    a1.h[1] = *reinterpret_cast<const v8bf*>(aRow + k0 + 40);
#pragma unroll
    for (int j = 0; j < TN; ++j)
      b1[j] = *reinterpret_cast<const v16bf*>(bRow[j] + k0 + 32);
#pragma unroll
    for (int j = 0; j < TN; ++j)
      acc[j] = __builtin_amdgcn_wmma_f32_16x16x32_bf16(
          false, a0.v, false, b0[j], (short)0, acc[j], false, false);

    if (k0 + 64 < DM) {  // compile-time after unroll
      // prefetch chunk k0+64 while chunk k0+32 computes
      a0.h[0] = *reinterpret_cast<const v8bf*>(aRow + k0 + 64);
      a0.h[1] = *reinterpret_cast<const v8bf*>(aRow + k0 + 72);
#pragma unroll
      for (int j = 0; j < TN; ++j)
        b0[j] = *reinterpret_cast<const v16bf*>(bRow[j] + k0 + 64);
    }
#pragma unroll
    for (int j = 0; j < TN; ++j)
      acc[j] = __builtin_amdgcn_wmma_f32_16x16x32_bf16(
          false, a1.v, false, b1[j], (short)0, acc[j], false, false);
  }

  const int mBase = mt * 16 + ((lane >> 4) << 3);
  const int nLo   = lane & 15;
#pragma unroll
  for (int j = 0; j < TN; ++j) {
    const int n = (ntB + j) * 16 + nLo;
    const float bv = bias[n];
#pragma unroll
    for (int r = 0; r < 8; ++r) {
      const int m = mBase + r;
      const float v = acc[j][r] + bv;
      if (MODE == 0) {
        Cf[(size_t)m * N + n] = v;
      } else {
        const int b = m / SS;           // batch
        const int s = m - b * SS;       // spatial
        const int h = n >> 5;           // head
        const int c = n & 31;           // channel
        Cb[(((size_t)(b * NHEADS + h) * SS + s) << 5) + c] = f2bf(v);
      }
    }
  }
}

// ---------------------------------------------------------------- sampling
// one wave per (b,q,h); one lane per channel (d = 32 == wave32)
__global__ __launch_bounds__(256) void deform_sample(
    const float* __restrict__ ref,     // [B*Q, 2]
    const float* __restrict__ logits,  // [B*Q, 96]: [0:64)=offsets, [64:96)=attn
    const u16* __restrict__ val,       // [B*NH, S, 32] bf16
    u16* __restrict__ samp) {          // [B*Q, 256] bf16
  const int gw   = (blockIdx.x * blockDim.x + threadIdx.x) >> 5;
  const int lane = threadIdx.x & 31;
  if (gw >= BB * QQ * NHEADS) return;
  const int h  = gw & (NHEADS - 1);
  const int bq = gw >> 3;
  const int b  = bq / QQ;

  const float rx = ref[bq * 2 + 0];
  const float ry = ref[bq * 2 + 1];
  const float* lrow = logits + (size_t)bq * NOA;

  float l[NPTS];
#pragma unroll
  for (int p = 0; p < NPTS; ++p) l[p] = lrow[64 + h * NPTS + p];
  const float mx = fmaxf(fmaxf(l[0], l[1]), fmaxf(l[2], l[3]));
  float e[NPTS], sum = 0.f;
#pragma unroll
  for (int p = 0; p < NPTS; ++p) { e[p] = __expf(l[p] - mx); sum += e[p]; }
  const float inv = 1.f / sum;

  const u16* vbase = val + ((size_t)(b * NHEADS + h) * SS) * HDIM + lane;

  float acc = 0.f;
#pragma unroll
  for (int p = 0; p < NPTS; ++p) {
    const float ox = lrow[h * (NPTS * 2) + p * 2 + 0];
    const float oy = lrow[h * (NPTS * 2) + p * 2 + 1];
    // loc = ref + off/[W,H]; pixel coord = loc*W - 0.5  ->  rx*W + ox - 0.5
    const float x = rx * (float)WW + ox - 0.5f;
    const float y = ry * (float)HH + oy - 0.5f;
    const float x0f = floorf(x), y0f = floorf(y);
    const int x0 = (int)x0f, y0 = (int)y0f;
    const float wx1 = x - x0f, wy1 = y - y0f;
    const float wx0 = 1.f - wx1, wy0 = 1.f - wy1;
    float s = 0.f;
#pragma unroll
    for (int cy = 0; cy < 2; ++cy) {
      const int yi = y0 + cy;
      const float wy = cy ? wy1 : wy0;
      if (yi < 0 || yi > HH - 1) continue;   // wave-uniform
#pragma unroll
      for (int cx = 0; cx < 2; ++cx) {
        const int xi = x0 + cx;
        const float wx = cx ? wx1 : wx0;
        if (xi < 0 || xi > WW - 1) continue; // wave-uniform
        s += wy * wx * bf2f(vbase[(size_t)(yi * WW + xi) * HDIM]);
      }
    }
    acc += (e[p] * inv) * s;
  }
  samp[(size_t)bq * DM + h * HDIM + lane] = f2bf(acc);
}

// ---------------------------------------------------------------- launch
extern "C" void kernel_launch(void* const* d_in, const int* in_sizes, int n_in,
                              void* d_out, int out_size, void* d_ws, size_t ws_size,
                              hipStream_t stream) {
  const float* hs    = (const float*)d_in[0];   // [B,Q,256]
  const float* enc   = (const float*)d_in[1];   // [B,S,256]
  const float* ref   = (const float*)d_in[2];   // [B,Q,1,2]
  // d_in[3] spatial_shapes: static, unused
  const float* Woff  = (const float*)d_in[4];   // [64,256]
  const float* boff  = (const float*)d_in[5];   // [64]
  const float* Wattn = (const float*)d_in[6];   // [32,256]
  const float* battn = (const float*)d_in[7];   // [32]
  const float* Wval  = (const float*)d_in[8];   // [256,256]
  const float* bval  = (const float*)d_in[9];   // [256]
  const float* Wout  = (const float*)d_in[10];  // [256,256]
  const float* bout  = (const float*)d_in[11];  // [256]
  float* out = (float*)d_out;
  (void)in_sizes; (void)n_in; (void)out_size; (void)ws_size;

  char* ws = (char*)d_ws;
  size_t off = 0;
  auto carve = [&](size_t bytes) -> char* {
    char* p = ws + off;
    off = (off + bytes + 255) & ~(size_t)255;
    return p;
  };
  const size_t MD = (size_t)MROWS * DM;
  u16*   hsb    = (u16*)carve(MD * 2);
  u16*   encb   = (u16*)carve(MD * 2);
  u16*   valb   = (u16*)carve(MD * 2);        // [B,NH,S,32] bf16
  u16*   sampb  = (u16*)carve(MD * 2);        // [B*Q,256] bf16
  float* logits = (float*)carve((size_t)MROWS * NOA * 4);
  u16*   Wvalb  = (u16*)carve((size_t)DM * DM * 2);
  u16*   Woab   = (u16*)carve((size_t)NOA * DM * 2);
  u16*   Woutb  = (u16*)carve((size_t)DM * DM * 2);
  float* boa    = (float*)carve(NOA * 4);

  convert_f32_bf16<<<2048, 256, 0, stream>>>(hs, hsb, (int)MD);
  convert_f32_bf16<<<2048, 256, 0, stream>>>(enc, encb, (int)MD);
  convert_f32_bf16<<<256, 256, 0, stream>>>(Wval, Wvalb, DM * DM);
  convert_f32_bf16<<<64, 256, 0, stream>>>(Woff, Woab, 64 * DM);
  convert_f32_bf16<<<32, 256, 0, stream>>>(Wattn, Woab + 64 * DM, 32 * DM);
  convert_f32_bf16<<<256, 256, 0, stream>>>(Wout, Woutb, DM * DM);
  pack_bias_96<<<1, 128, 0, stream>>>(boff, battn, boa);

  const dim3 blk(32, 1, 1);
  // 1) value = enc @ Wval^T + bval  -> bf16 [B,NH,S,32]
  gemm_bf16_wmma<4, 1><<<dim3(MROWS / 16, DM / 64), blk, 0, stream>>>(
      encb, Wvalb, bval, nullptr, valb, DM);
  // 2) [offsets | attn logits] = hs @ [Woff;Wattn]^T + [boff;battn] -> f32 [M,96]
  gemm_bf16_wmma<2, 0><<<dim3(MROWS / 16, NOA / 32), blk, 0, stream>>>(
      hsb, Woab, boa, logits, nullptr, NOA);
  // 3) deformable bilinear sampling + softmax-weighted sum -> bf16 [M,256]
  const int nwaves = BB * QQ * NHEADS;               // 320000
  deform_sample<<<(nwaves * 32) / 256, 256, 0, stream>>>(ref, logits, valb, sampb);
  // 4) out = sampled @ Wout^T + bout -> f32 d_out
  gemm_bf16_wmma<4, 0><<<dim3(MROWS / 16, DM / 64), blk, 0, stream>>>(
      sampb, Woutb, bout, out, nullptr, DM);
}
